// PIKANNetwork_32976758899382
// MI455X (gfx1250) — matrix operand order repaired and checked by reading, compiled, not run on hardware
//
#include <hip/hip_runtime.h>

typedef __attribute__((ext_vector_type(16))) _Float16 v16h;
typedef __attribute__((ext_vector_type(8)))  _Float16 v8h;
typedef __attribute__((ext_vector_type(8)))  float    v8f;

#define NBASIS 10
#define EPSF   1e-8f

// ---------------------------------------------------------------------------
// Prep kernels: pack fp32 params into fp16 GEMM operands (re-done every call).
// ---------------------------------------------------------------------------

__global__ __launch_bounds__(256) void prep_wq_kernel(
    const float* __restrict__ ocent, const float* __restrict__ oscal,
    _Float16* __restrict__ Wq, int D, int dout, int N1, int total)
{
  int idx = blockIdx.x * 256 + threadIdx.x;
  if (idx >= total) return;
  int k = idx / N1;
  int n = idx - k * N1;
  float v = 0.f;
  if (n < dout) {
    if (k < D) {
      float s = oscal[(size_t)n * D + k] + EPSF;
      v = 1.f / (s * s);
    } else if (k < 2 * D) {
      int kk = k - D;
      float s = oscal[(size_t)n * D + kk] + EPSF;
      float is2 = 1.f / (s * s);
      v = -2.f * ocent[(size_t)n * D + kk] * is2;
    }
  }
  Wq[idx] = (_Float16)v;
}

__global__ __launch_bounds__(64) void prep_c2_kernel(
    const float* __restrict__ ocent, const float* __restrict__ oscal,
    float* __restrict__ c2, int D, int dout, int N1)
{
  int n = blockIdx.x * 64 + threadIdx.x;
  if (n >= N1) return;
  float acc = 0.f;
  if (n < dout) {
    for (int k = 0; k < D; ++k) {
      float s = oscal[(size_t)n * D + k] + EPSF;
      float is2 = 1.f / (s * s);
      float c = ocent[(size_t)n * D + k];
      acc += c * c * is2;
    }
  }
  c2[n] = acc;
}

__global__ __launch_bounds__(256) void prep_wl_kernel(
    const float* __restrict__ linW, _Float16* __restrict__ WL,
    int dout, int N2, int total)
{
  int idx = blockIdx.x * 256 + threadIdx.x;
  if (idx >= total) return;
  int k = idx / N2;
  int n = idx - k * N2;
  float v = (k < dout && n < dout) ? linW[(size_t)n * dout + k] : 0.f;
  WL[idx] = (_Float16)v;
}

// ---------------------------------------------------------------------------
// Fused KA layer, fully shape-specialized (4 instantiations).
// 16 batch rows per workgroup, 8 waves, K split across 8/ntiles waves per
// N-tile with LDS partial-tile reduction -> all waves issue WMMA every layer.
// ---------------------------------------------------------------------------
template <int DIN, int D, int K1PAD, int DOUT, int N1, int K2, int N2>
__global__ __launch_bounds__(256) void ka_layer_kernel(
    const float* __restrict__ Hin,
    const float* __restrict__ icent,
    const float* __restrict__ iscale,
    const float* __restrict__ iw,
    const _Float16* __restrict__ Wq,
    const float* __restrict__ c2sum,
    const float* __restrict__ outer_w,
    const float* __restrict__ theta_p,
    const float* __restrict__ kappa_p,
    const _Float16* __restrict__ WL,
    const float* __restrict__ lin_b,
    float* __restrict__ Hout)
{
  extern __shared__ char smem_raw[];
  _Float16* Z   = (_Float16*)smem_raw;                               // [16][K1PAD]
  float*    Xs  = (float*)(smem_raw + (size_t)16 * K1PAD * 2);       // [16][DIN]
  float*    ssq = Xs + 16 * DIN;                                     // [16]
  float*    rm  = ssq + 16;                                          // [16]
  float*    Qp  = rm + 16;                                           // [8][16][16]
  _Float16* G   = (_Float16*)(Qp + 8 * 256);                         // [16][K2]

  const int tid  = threadIdx.x;
  const int lane = tid & 31;
  // force wave id into an SGPR: scalar loop control, no exec masking around WMMA
  const int wave = __builtin_amdgcn_readfirstlane(tid >> 5);
  const int row0 = blockIdx.x * 16;
  const int hi   = (lane >> 4) & 1;
  const int mrow = lane & 15;

  // ---- stage 0: coalesced copy of the contiguous 16 x DIN input strip ----
  for (int i = tid; i < 16 * DIN; i += 256)
    Xs[i] = Hin[(size_t)row0 * DIN + i];

  // ---- init LDS ----
  for (int i = tid; i < 16 * K2; i += 256) G[i] = (_Float16)0.f;
  if (tid < 16) ssq[tid] = 0.f;
  constexpr int PAD = K1PAD - 2 * D;
  if (PAD > 0) {
    for (int i = tid; i < 16 * (PAD > 0 ? PAD : 1); i += 256) {
      int r = i / (PAD > 0 ? PAD : 1), j = i - r * (PAD > 0 ? PAD : 1);
      Z[r * K1PAD + 2 * D + j] = (_Float16)0.f;
    }
  }
  __syncthreads();

  // ---- stage 1: inner RBF expansion ----
  for (int t = tid; t < 16 * D; t += 256) {
    int r = t / D, k = t - r * D;
    int d = k / NBASIS, j = k - d * NBASIS;
    float xv = Xs[r * DIN + d];
    float c  = icent[d * NBASIS + j];
    float s  = iscale[d * NBASIS + j] + EPSF;
    float u  = (xv - c) / s;
    float g  = __expf(-0.5f * u * u);
    float z  = g * iw[d * NBASIS + j];
    Z[r * K1PAD + k]     = (_Float16)(z * z);
    Z[r * K1PAD + D + k] = (_Float16)z;
    atomicAdd(&ssq[r], z * z);
  }
  __syncthreads();

  if (tid < 16) {
    float z0 = (float)Z[tid * K1PAD + D + 0];
    float z1 = (float)Z[tid * K1PAD + D + 1];
    rm[tid] = theta_p[0] * z0 * z1 + kappa_p[0] * (2.f * z0 * z0 - ssq[tid]);
  }
  __syncthreads();

  // ---- stage 2: q-GEMM (Zcat @ Wq), K split across waves ----
  {
    constexpr int NT  = N1 >> 4;      // 1, 4 or 8 N-tiles
    constexpr int WPT = 8 / NT;       // waves per tile
    constexpr int NC  = K1PAD >> 5;   // 32-wide K chunks
    const int tile = wave / WPT;
    const int sub  = wave - tile * WPT;

    v8f acc = {};
#pragma unroll 4
    for (int c = sub; c < NC; c += WPT) {
      const int k0 = c << 5;
      const _Float16* zr = Z + mrow * K1PAD + k0 + hi * 8;
      v8h alo = *(const v8h*)(zr);
      v8h ahi = *(const v8h*)(zr + 16);
      v16h a;
#pragma unroll
      for (int i = 0; i < 8; ++i) { a[i] = alo[i]; a[8 + i] = ahi[i]; }
      const _Float16* br = Wq + (size_t)(k0 + hi * 8) * N1 + tile * 16 + mrow;
      v16h b;
#pragma unroll
      for (int i = 0; i < 8; ++i) { b[i] = br[(size_t)i * N1]; b[8 + i] = br[(size_t)(16 + i) * N1]; }
      acc = __builtin_amdgcn_wmma_f32_16x16x32_f16(false, a, false, b,
                                                   (short)0, acc, false, false);
    }
#pragma unroll
    for (int r8 = 0; r8 < 8; ++r8)
      Qp[wave * 256 + (r8 + hi * 8) * 16 + mrow] = acc[r8];
    __syncthreads();

    // gaussian + theta/kappa epilogue, all 256 threads
    for (int e = tid; e < 16 * N1; e += 256) {
      int row = e / N1, col = e - row * N1;
      int t = col >> 4, ci = col & 15;
      float q = 0.f;
#pragma unroll
      for (int i = 0; i < WPT; ++i)
        q += Qp[(t * WPT + i) * 256 + row * 16 + ci];
      q += c2sum[col];
      float go = __expf(-0.5f * q + rm[row]);
      float ow = (col < DOUT) ? outer_w[col] : 0.f;
      G[row * K2 + col] = (_Float16)(go * ow);
    }
  }
  __syncthreads();

  // ---- stage 3: linear GEMM (G @ WL), same K-split, bias + relu ----
  {
    constexpr int NT  = N2 >> 4;
    constexpr int WPT = 8 / NT;
    constexpr int NC  = K2 >> 5;
    const int tile = wave / WPT;
    const int sub  = wave - tile * WPT;

    v8f acc = {};
#pragma unroll 4
    for (int c = sub; c < NC; c += WPT) {
      const int k0 = c << 5;
      const _Float16* gr = G + mrow * K2 + k0 + hi * 8;
      v8h alo = *(const v8h*)(gr);
      v8h ahi = *(const v8h*)(gr + 16);
      v16h a;
#pragma unroll
      for (int i = 0; i < 8; ++i) { a[i] = alo[i]; a[8 + i] = ahi[i]; }
      const _Float16* br = WL + (size_t)(k0 + hi * 8) * N2 + tile * 16 + mrow;
      v16h b;
#pragma unroll
      for (int i = 0; i < 8; ++i) { b[i] = br[(size_t)i * N2]; b[8 + i] = br[(size_t)(16 + i) * N2]; }
      acc = __builtin_amdgcn_wmma_f32_16x16x32_f16(false, a, false, b,
                                                   (short)0, acc, false, false);
    }
#pragma unroll
    for (int r8 = 0; r8 < 8; ++r8)
      Qp[wave * 256 + (r8 + hi * 8) * 16 + mrow] = acc[r8];
    __syncthreads();

    for (int e = tid; e < 16 * DOUT; e += 256) {
      int row = e / DOUT, col = e - row * DOUT;
      int t = col >> 4, ci = col & 15;
      float v = 0.f;
#pragma unroll
      for (int i = 0; i < WPT; ++i)
        v += Qp[(t * WPT + i) * 256 + row * 16 + ci];
      v += lin_b[col];
      Hout[(size_t)(row0 + row) * DOUT + col] = v > 0.f ? v : 0.f;
    }
  }
}

// ---------------------------------------------------------------------------
// Host side
// ---------------------------------------------------------------------------
static inline size_t layer_shmem(int K1pad, int din, int K2) {
  return (size_t)16 * K1pad * 2 + (size_t)16 * din * 4 + 32 * sizeof(float)
       + 8 * 256 * sizeof(float) + (size_t)16 * K2 * 2;
}

extern "C" void kernel_launch(void* const* d_in, const int* in_sizes, int n_in,
                              void* d_out, int out_size, void* d_ws, size_t ws_size,
                              hipStream_t stream) {
  (void)in_sizes; (void)n_in; (void)out_size; (void)ws_size;
  const int B = 16384;
  const int dims[5] = {4, 64, 128, 64, 1};

  char* ws = (char*)d_ws;
  size_t off = 0;
  auto alloc = [&](size_t bytes) -> void* {
    off = (off + 255) & ~(size_t)255;
    void* p = ws + off;
    off += bytes;
    return p;
  };

  float* Hb0 = (float*)alloc((size_t)B * 64  * sizeof(float));
  float* Hb1 = (float*)alloc((size_t)B * 128 * sizeof(float));

  _Float16* Wq[4]; float* C2[4]; _Float16* WL[4];
  int Dv[4], K1v[4], N1v[4], K2v[4], N2v[4];
  for (int l = 0; l < 4; ++l) {
    int din = dims[l], dout = dims[l + 1];
    int D = din * NBASIS;
    int K1pad = ((2 * D + 31) / 32) * 32;
    int N1 = ((dout + 15) / 16) * 16;
    int K2 = ((dout + 31) / 32) * 32;
    int N2 = N1;
    Dv[l] = D; K1v[l] = K1pad; N1v[l] = N1; K2v[l] = K2; N2v[l] = N2;
    Wq[l] = (_Float16*)alloc((size_t)K1pad * N1 * 2);
    C2[l] = (float*)alloc((size_t)N1 * 4);
    WL[l] = (_Float16*)alloc((size_t)K2 * N2 * 2);
  }

  for (int l = 0; l < 4; ++l) {
    int base = 1 + 10 * l;
    const float* ocent = (const float*)d_in[base + 3];
    const float* oscal = (const float*)d_in[base + 4];
    const float* linW  = (const float*)d_in[base + 8];
    int dout = dims[l + 1], D = Dv[l];
    int tot1 = K1v[l] * N1v[l];
    prep_wq_kernel<<<(tot1 + 255) / 256, 256, 0, stream>>>(ocent, oscal, Wq[l], D, dout, N1v[l], tot1);
    prep_c2_kernel<<<(N1v[l] + 63) / 64, 64, 0, stream>>>(ocent, oscal, C2[l], D, dout, N1v[l]);
    int tot2 = K2v[l] * N2v[l];
    prep_wl_kernel<<<(tot2 + 255) / 256, 256, 0, stream>>>(linW, WL[l], dout, N2v[l], tot2);
  }

  // layer chain: x -> Hb0 -> Hb1 -> Hb0 -> d_out
  const float* X = (const float*)d_in[0];
  const void* A[4][8];
  for (int l = 0; l < 4; ++l) {
    int base = 1 + 10 * l;
    A[l][0] = d_in[base + 0]; A[l][1] = d_in[base + 1]; A[l][2] = d_in[base + 2];
    A[l][3] = d_in[base + 5]; A[l][4] = d_in[base + 6]; A[l][5] = d_in[base + 7];
    A[l][6] = d_in[base + 9];
  }
  const int nb = B / 16;

  ka_layer_kernel<4, 40, 96, 64, 64, 64, 64>
      <<<nb, 256, layer_shmem(96, 4, 64), stream>>>(
      X, (const float*)A[0][0], (const float*)A[0][1], (const float*)A[0][2],
      Wq[0], C2[0], (const float*)A[0][3], (const float*)A[0][4], (const float*)A[0][5],
      WL[0], (const float*)A[0][6], Hb0);

  ka_layer_kernel<64, 640, 1280, 128, 128, 128, 128>
      <<<nb, 256, layer_shmem(1280, 64, 128), stream>>>(
      Hb0, (const float*)A[1][0], (const float*)A[1][1], (const float*)A[1][2],
      Wq[1], C2[1], (const float*)A[1][3], (const float*)A[1][4], (const float*)A[1][5],
      WL[1], (const float*)A[1][6], Hb1);

  ka_layer_kernel<128, 1280, 2560, 64, 64, 64, 64>
      <<<nb, 256, layer_shmem(2560, 128, 64), stream>>>(
      Hb1, (const float*)A[2][0], (const float*)A[2][1], (const float*)A[2][2],
      Wq[2], C2[2], (const float*)A[2][3], (const float*)A[2][4], (const float*)A[2][5],
      WL[2], (const float*)A[2][6], Hb0);

  ka_layer_kernel<64, 640, 1280, 1, 16, 32, 16>
      <<<nb, 256, layer_shmem(1280, 64, 32), stream>>>(
      Hb0, (const float*)A[3][0], (const float*)A[3][1], (const float*)A[3][2],
      Wq[3], C2[3], (const float*)A[3][3], (const float*)A[3][4], (const float*)A[3][5],
      WL[3], (const float*)A[3][6], (float*)d_out);
}
